// BNN_VGG7_32461362823713
// MI455X (gfx1250) — 8-bit matrix operand order repaired and checked by reading, compiled, not run on hardware
//
#include <hip/hip_runtime.h>
#include <cstdint>


typedef __attribute__((ext_vector_type(8))) int v8i;

#define BN_EPS 1e-5f

__device__ __forceinline__ int iclampi(int v, int lo, int hi) {
    return v < lo ? lo : (v > hi ? hi : v);
}

__device__ __forceinline__ v8i load_a_frag(const int8_t* p, int khalf) {
    const int8_t* q = p + (khalf << 4);
    int2 a01 = *(const int2*)(q);
    int2 a23 = *(const int2*)(q + 8);
    int2 a45 = *(const int2*)(q + 32);
    int2 a67 = *(const int2*)(q + 40);
    v8i a = {a01.x, a01.y, a23.x, a23.y, a45.x, a45.y, a67.x, a67.y};
    return a;
}

__device__ __forceinline__ v8i load_b_frag(const int8_t* p, int khalf) {
    const int8_t* q = p + (khalf << 4);
    int4 lo = *(const int4*)(q);
    int4 hi = *(const int4*)(q + 32);
    v8i b = {lo.x, lo.y, lo.z, lo.w, hi.x, hi.y, hi.z, hi.w};
    return b;
}


__global__ void prep_convw_kernel(const float* __restrict__ w,
                                  int8_t* __restrict__ wb,
                                  int Cout, int Cin) {
    int idx = blockIdx.x * blockDim.x + threadIdx.x;
    int total = Cout * 9 * Cin;
    if (idx >= total) return;
    int ci = idx % Cin;
    int t  = (idx / Cin) % 9;
    int co = idx / (9 * Cin);
    float v = w[(co * Cin + ci) * 9 + t];
    wb[idx] = (v >= 0.f) ? (int8_t)1 : (int8_t)-1;
}

__global__ void prep_fcw_kernel(const float* __restrict__ w,
                                int8_t* __restrict__ wb, int n) {
    int idx = blockIdx.x * blockDim.x + threadIdx.x;
    if (idx >= n) return;
    wb[idx] = (w[idx] >= 0.f) ? (int8_t)1 : (int8_t)-1;
}

__global__ void conv1_kernel(const float* __restrict__ x,
                             const float* __restrict__ w1,
                             const float* __restrict__ bn_g,
                             const float* __restrict__ bn_b,
                             const float* __restrict__ bn_m,
                             const float* __restrict__ bn_v,
                             int8_t* __restrict__ out) {
    int idx = blockIdx.x * blockDim.x + threadIdx.x;
    if (idx >= 128 * 32 * 32 * 128) return;
    int co = idx & 127;
    int r  = idx >> 7;
    int xx = r & 31; r >>= 5;
    int yy = r & 31; r >>= 5;
    int bidx = r;
    float acc = 0.f;
#pragma unroll
    for (int dy = 0; dy < 3; ++dy) {
        int ys = iclampi(yy + dy - 1, 0, 31);
#pragma unroll
        for (int dx = 0; dx < 3; ++dx) {
            int xs = iclampi(xx + dx - 1, 0, 31);
#pragma unroll
            for (int ci = 0; ci < 3; ++ci) {
                float wv = w1[(co * 3 + ci) * 9 + dy * 3 + dx];
                float xv = x[((bidx * 3 + ci) * 32 + ys) * 32 + xs];
                acc += (wv >= 0.f) ? xv : -xv;
            }
        }
    }
    float inv = bn_g[co] * rsqrtf(bn_v[co] + BN_EPS);
    float y = (acc - bn_m[co]) * inv + bn_b[co];
    out[idx] = (y >= 0.f) ? (int8_t)1 : (int8_t)-1;
}

__global__ __launch_bounds__(32)
void binconv_wmma_kernel(const int8_t* __restrict__ act,
                         const int8_t* __restrict__ wb,
                         int8_t* __restrict__ out,
                         const float* __restrict__ bn_g,
                         const float* __restrict__ bn_b,
                         const float* __restrict__ bn_m,
                         const float* __restrict__ bn_v,
                         int H, int W, int lw, int lhw,
                         int Cin, int Cout) {
    const int lane  = threadIdx.x & 31;
    const int mtile = blockIdx.x;
    const int HW = H * W;

    const int mrow  = lane & 15;
    const int khalf = lane >> 4;
    const int pos   = mtile * 16 + mrow;
    const int bidx  = pos >> lhw;
    const int rem   = pos & (HW - 1);
    const int oy    = rem >> lw;
    const int ox    = rem & (W - 1);

    const int ncol0 = blockIdx.y * 64 + (lane & 15);
    const size_t wstride = (size_t)9 * Cin;
    const int8_t* wrow0 = wb + (size_t)ncol0 * wstride;

    v8i acc0 = {}, acc1 = {}, acc2 = {}, acc3 = {};
    const int nchunks = Cin >> 6;
    for (int t = 0; t < 9; ++t) {
        int dy = t / 3 - 1, dx = t % 3 - 1;
        int ys = iclampi(oy + dy, 0, H - 1);
        int xs = iclampi(ox + dx, 0, W - 1);
        const int8_t* ap = act + ((size_t)bidx * HW + ys * W + xs) * Cin;
        if (t < 8) {
            int t2 = t + 1;
            int ys2 = iclampi(oy + t2 / 3 - 1, 0, H - 1);
            int xs2 = iclampi(ox + t2 % 3 - 1, 0, W - 1);
            __builtin_prefetch(act + ((size_t)bidx * HW + ys2 * W + xs2) * Cin, 0, 0);
        }
        const int8_t* wp = wrow0 + t * Cin;
        for (int cc = 0; cc < nchunks; ++cc) {
            v8i a = load_a_frag(ap + cc * 64, khalf);
            v8i b0 = load_b_frag(wp + cc * 64, khalf);
            acc0 = __builtin_amdgcn_wmma_i32_16x16x64_iu8(true, a, true, b0, acc0, false, false);
            v8i b1 = load_b_frag(wp + 16 * wstride + cc * 64, khalf);
            acc1 = __builtin_amdgcn_wmma_i32_16x16x64_iu8(true, a, true, b1, acc1, false, false);
            v8i b2 = load_b_frag(wp + 32 * wstride + cc * 64, khalf);
            acc2 = __builtin_amdgcn_wmma_i32_16x16x64_iu8(true, a, true, b2, acc2, false, false);
            v8i b3 = load_b_frag(wp + 48 * wstride + cc * 64, khalf);
            acc3 = __builtin_amdgcn_wmma_i32_16x16x64_iu8(true, a, true, b3, acc3, false, false);
        }
    }

    v8i accs[4] = {acc0, acc1, acc2, acc3};
#pragma unroll
    for (int j = 0; j < 4; ++j) {
        int ncol = ncol0 + j * 16;
        float inv = bn_g[ncol] * rsqrtf(bn_v[ncol] + BN_EPS);
        float mu  = bn_m[ncol];
        float bbv = bn_b[ncol];
#pragma unroll
        for (int r = 0; r < 8; ++r) {
            int m = r + (khalf << 3);
            int p = mtile * 16 + m;
            float y = ((float)accs[j][r] - mu) * inv + bbv;
            out[(size_t)p * Cout + ncol] = (y >= 0.f) ? (int8_t)1 : (int8_t)-1;
        }
    }
}

__global__ void maxpool_kernel(const int8_t* __restrict__ in,
                               int8_t* __restrict__ out,
                               int B, int H, int W, int C) {
    int Ho = H >> 1, Wo = W >> 1;
    int total = B * Ho * Wo * C;
    int idx = blockIdx.x * blockDim.x + threadIdx.x;
    if (idx >= total) return;
    int c  = idx % C;
    int r  = idx / C;
    int xo = r % Wo; r /= Wo;
    int yo = r % Ho; r /= Ho;
    int b  = r;
    const int8_t* p = in + (((size_t)b * H + 2 * yo) * W + 2 * xo) * C + c;
    int8_t v0 = p[0], v1 = p[C], v2 = p[(size_t)W * C], v3 = p[(size_t)W * C + C];
    int8_t m01 = v0 > v1 ? v0 : v1;
    int8_t m23 = v2 > v3 ? v2 : v3;
    out[idx] = m01 > m23 ? m01 : m23;
}

__global__ void repack_kernel(const int8_t* __restrict__ in,
                              int8_t* __restrict__ out) {
    int idx = blockIdx.x * blockDim.x + threadIdx.x;
    if (idx >= 128 * 8192) return;
    int k = idx & 8191;
    int b = idx >> 13;
    int c = k >> 4;
    int r = k & 15;
    out[idx] = in[((size_t)b * 16 + r) * 512 + c];
}

__global__ __launch_bounds__(32)
void bingemm_wmma_kernel(const int8_t* __restrict__ A,
                         const int8_t* __restrict__ Wb,
                         int8_t* __restrict__ out,
                         const float* __restrict__ bn_g,
                         const float* __restrict__ bn_b,
                         const float* __restrict__ bn_m,
                         const float* __restrict__ bn_v,
                         int K, int N) {
    const int lane  = threadIdx.x & 31;
    const int mtile = blockIdx.x;

    const int mrow  = lane & 15;
    const int khalf = lane >> 4;
    const int ncol0 = blockIdx.y * 64 + (lane & 15);

    const int8_t* ap = A + (size_t)(mtile * 16 + mrow) * K;
    const int8_t* wp = Wb + (size_t)ncol0 * K;

    v8i acc0 = {}, acc1 = {}, acc2 = {}, acc3 = {};
    const int nchunks = K >> 6;
    for (int cc = 0; cc < nchunks; ++cc) {
        v8i a = load_a_frag(ap + cc * 64, khalf);
        v8i b0 = load_b_frag(wp + cc * 64, khalf);
        acc0 = __builtin_amdgcn_wmma_i32_16x16x64_iu8(true, a, true, b0, acc0, false, false);
        v8i b1 = load_b_frag(wp + (size_t)16 * K + cc * 64, khalf);
        acc1 = __builtin_amdgcn_wmma_i32_16x16x64_iu8(true, a, true, b1, acc1, false, false);
        v8i b2 = load_b_frag(wp + (size_t)32 * K + cc * 64, khalf);
        acc2 = __builtin_amdgcn_wmma_i32_16x16x64_iu8(true, a, true, b2, acc2, false, false);
        v8i b3 = load_b_frag(wp + (size_t)48 * K + cc * 64, khalf);
        acc3 = __builtin_amdgcn_wmma_i32_16x16x64_iu8(true, a, true, b3, acc3, false, false);
    }

    v8i accs[4] = {acc0, acc1, acc2, acc3};
#pragma unroll
    for (int j = 0; j < 4; ++j) {
        int ncol = ncol0 + j * 16;
        float inv = bn_g[ncol] * rsqrtf(bn_v[ncol] + BN_EPS);
        float mu  = bn_m[ncol];
        float bbv = bn_b[ncol];
#pragma unroll
        for (int r = 0; r < 8; ++r) {
            int m = r + (khalf << 3);
            int p = mtile * 16 + m;
            float y = ((float)accs[j][r] - mu) * inv + bbv;
            out[(size_t)p * N + ncol] = (y >= 0.f) ? (int8_t)1 : (int8_t)-1;
        }
    }
}

__global__ void fc2_kernel(const int8_t* __restrict__ a,
                           const int8_t* __restrict__ wb,
                           const float* __restrict__ scale,
                           float* __restrict__ out) {
    int idx = blockIdx.x * blockDim.x + threadIdx.x;
    if (idx >= 128 * 10) return;
    int b = idx / 10, j = idx % 10;
    const unsigned* ap = (const unsigned*)(a + (size_t)b * 1024);
    const unsigned* wp = (const unsigned*)(wb + (size_t)j * 1024);
    int ndiff = 0;
    for (int k = 0; k < 256; ++k) {
        unsigned x = ap[k] ^ wp[k];
        ndiff += __popc((x >> 1) & 0x01010101u);
    }
    out[idx] = scale[0] * (float)(1024 - 2 * ndiff);
}


extern "C" void kernel_launch(void* const* d_in, const int* in_sizes, int n_in,
                              void* d_out, int out_size, void* d_ws, size_t ws_size,
                              hipStream_t stream) {
    (void)in_sizes; (void)n_in; (void)out_size; (void)ws_size;
    const float* x    = (const float*)d_in[0];
    const float* w1   = (const float*)d_in[1];
    const float* w2   = (const float*)d_in[2];
    const float* w3   = (const float*)d_in[3];
    const float* w4   = (const float*)d_in[4];
    const float* w5   = (const float*)d_in[5];
    const float* w6   = (const float*)d_in[6];
    const float* fc1w = (const float*)d_in[7];
    const float* fc2w = (const float*)d_in[8];
    auto bn = [&](int i, int j) { return (const float*)d_in[9 + (i - 1) * 4 + j]; };
    const float* scale = (const float*)d_in[37];

    char* ws = (char*)d_ws;
    size_t off = 0;
    auto alloc = [&](size_t bytes) {
        size_t r = off;
        off += (bytes + 255) & ~(size_t)255;
        return r;
    };
    int8_t* wb2   = (int8_t*)(ws + alloc(128 * 9 * 128));
    int8_t* wb3   = (int8_t*)(ws + alloc(256 * 9 * 128));
    int8_t* wb4   = (int8_t*)(ws + alloc(256 * 9 * 256));
    int8_t* wb5   = (int8_t*)(ws + alloc(512 * 9 * 256));
    int8_t* wb6   = (int8_t*)(ws + alloc(512 * 9 * 512));
    int8_t* wbfc1 = (int8_t*)(ws + alloc((size_t)1024 * 8192));
    int8_t* wbfc2 = (int8_t*)(ws + alloc(10 * 1024));
    int8_t* bufA  = (int8_t*)(ws + alloc((size_t)128 * 32 * 32 * 128));
    int8_t* bufB  = (int8_t*)(ws + alloc((size_t)128 * 32 * 32 * 128));

    auto g1 = [](int n) { return (n + 255) / 256; };

    prep_convw_kernel<<<g1(128 * 9 * 128), 256, 0, stream>>>(w2, wb2, 128, 128);
    prep_convw_kernel<<<g1(256 * 9 * 128), 256, 0, stream>>>(w3, wb3, 256, 128);
    prep_convw_kernel<<<g1(256 * 9 * 256), 256, 0, stream>>>(w4, wb4, 256, 256);
    prep_convw_kernel<<<g1(512 * 9 * 256), 256, 0, stream>>>(w5, wb5, 512, 256);
    prep_convw_kernel<<<g1(512 * 9 * 512), 256, 0, stream>>>(w6, wb6, 512, 512);
    prep_fcw_kernel<<<g1(1024 * 8192), 256, 0, stream>>>(fc1w, wbfc1, 1024 * 8192);
    prep_fcw_kernel<<<g1(10 * 1024), 256, 0, stream>>>(fc2w, wbfc2, 10 * 1024);

    conv1_kernel<<<g1(128 * 32 * 32 * 128), 256, 0, stream>>>(
        x, w1, bn(1, 0), bn(1, 1), bn(1, 2), bn(1, 3), bufA);

    binconv_wmma_kernel<<<dim3(128 * 32 * 32 / 16, 128 / 64), 32, 0, stream>>>(
        bufA, wb2, bufB, bn(2, 0), bn(2, 1), bn(2, 2), bn(2, 3),
        32, 32, 5, 10, 128, 128);
    maxpool_kernel<<<g1(128 * 16 * 16 * 128), 256, 0, stream>>>(bufB, bufA, 128, 32, 32, 128);

    binconv_wmma_kernel<<<dim3(128 * 16 * 16 / 16, 256 / 64), 32, 0, stream>>>(
        bufA, wb3, bufB, bn(3, 0), bn(3, 1), bn(3, 2), bn(3, 3),
        16, 16, 4, 8, 128, 256);
    binconv_wmma_kernel<<<dim3(128 * 16 * 16 / 16, 256 / 64), 32, 0, stream>>>(
        bufB, wb4, bufA, bn(4, 0), bn(4, 1), bn(4, 2), bn(4, 3),
        16, 16, 4, 8, 256, 256);
    maxpool_kernel<<<g1(128 * 8 * 8 * 256), 256, 0, stream>>>(bufA, bufB, 128, 16, 16, 256);

    binconv_wmma_kernel<<<dim3(128 * 8 * 8 / 16, 512 / 64), 32, 0, stream>>>(
        bufB, wb5, bufA, bn(5, 0), bn(5, 1), bn(5, 2), bn(5, 3),
        8, 8, 3, 6, 256, 512);
    binconv_wmma_kernel<<<dim3(128 * 8 * 8 / 16, 512 / 64), 32, 0, stream>>>(
        bufA, wb6, bufB, bn(6, 0), bn(6, 1), bn(6, 2), bn(6, 3),
        8, 8, 3, 6, 512, 512);
    maxpool_kernel<<<g1(128 * 4 * 4 * 512), 256, 0, stream>>>(bufB, bufA, 128, 8, 8, 512);

    repack_kernel<<<g1(128 * 8192), 256, 0, stream>>>(bufA, bufB);

    bingemm_wmma_kernel<<<dim3(128 / 16, 1024 / 64), 32, 0, stream>>>(
        bufB, wbfc1, bufA, bn(7, 0), bn(7, 1), bn(7, 2), bn(7, 3), 8192, 1024);

    fc2_kernel<<<g1(128 * 10), 256, 0, stream>>>(bufA, wbfc2, scale, (float*)d_out);
}
